// CirculantTemporalEncoding_61890478735595
// MI455X (gfx1250) — matrix operand order repaired and checked by reading, not compiled, not run on hardware
//
#include <hip/hip_runtime.h>
#include <hip/hip_bf16.h>

// ---------------------------------------------------------------------------
// CirculantTemporalEncoding on MI455X (gfx1250)
//
// Truncated 4096-pt DFT (keep 1024 bins) done as f16 WMMA GEMMs with f32
// accumulation:  Q[b, k, d] = sum_t W[k,t] * ts[b, t, d]
// Four products (cos/-sin basis x ts_a/ts_b) fused per workgroup so the
// per-(b,k) reductions over d (|Q|^2 sums and conj(Qa)*Qb) are computed
// straight out of the WMMA accumulators (lane butterfly + global atomics),
// avoiding any Q materialization. Final 1024-bin inverse DFT + flip/concat
// is a tiny VALU kernel using a rotation recurrence.
// ---------------------------------------------------------------------------

typedef _Float16 v16h __attribute__((ext_vector_type(16)));
typedef _Float16 v8h  __attribute__((ext_vector_type(8)));
typedef float    v8f  __attribute__((ext_vector_type(8)));

static constexpr int   BATCH  = 16;
static constexpr int   TLEN   = 4096;   // L == next_pow2(L) == Lp2
static constexpr int   DDIM   = 512;
static constexpr int   NBIN   = 1024;   // m retained bins
static constexpr float LMBDA  = 0.1f;
static constexpr float TWO_PI = 6.28318530717958647692f;

// ws layout (bytes):
//   [0,              8 MB)  : Wc  f16 [NBIN][TLEN]   cos(2*pi*k*t/4096)
//   [8 MB,          16 MB)  : Wm  f16 [NBIN][TLEN]  -sin(2*pi*k*t/4096)
//   [16 MB, 16 MB + 256 KB) : acc f32 [BATCH][4][NBIN]  (dena, denb, cr, ci)
static constexpr size_t WS_TBL_BYTES = (size_t)NBIN * TLEN * sizeof(_Float16); // 8 MB

// ------------------------------ twiddle tables -----------------------------
__global__ __launch_bounds__(256)
void cte_table_kernel(_Float16* __restrict__ Wc, _Float16* __restrict__ Wm) {
    int idx = blockIdx.x * 256 + threadIdx.x;      // 0 .. NBIN*TLEN-1
    int k = idx >> 12;                             // bin
    int t = idx & (TLEN - 1);                      // time
    float s, c;
    __sincosf((TWO_PI / (float)TLEN) * (float)((k * t) & (TLEN - 1)), &s, &c);
    Wc[idx] = (_Float16)c;
    Wm[idx] = (_Float16)(-s);                      // Qi = sum x * (-sin)
}

__global__ __launch_bounds__(256)
void cte_zero_kernel(float* __restrict__ p, int n) {
    int i = blockIdx.x * 256 + threadIdx.x;
    if (i < n) p[i] = 0.0f;
}

// --------------------- fused DFT GEMM + d-reduction ------------------------
// Block tile: BM=128 (k bins) x BN=64 (d) over K=TLEN in BK=32 steps.
// 8 waves (4 x 2), wave tile 32x32 -> 2x2 C frags per product, 4 products.
__global__ __launch_bounds__(256)
void cte_dft_corr_kernel(const float* __restrict__ tsa,
                         const float* __restrict__ tsb,
                         const _Float16* __restrict__ Wc,
                         const _Float16* __restrict__ Wm,
                         float* __restrict__ accg) {
    constexpr int BM = 128, BN = 64, BK = 32, PAD = 40; // 40 halves = 80 B rows

    __shared__ _Float16 As[2][BM][PAD];  // [part: cos/-sin][k row][kk]
    __shared__ _Float16 Bs[2][BN][PAD];  // [tensor a/b][d][kk]  (transposed)

    const int tid   = threadIdx.x;
    const int lane  = tid & 31;
    const int wave  = tid >> 5;
    const int wm    = wave >> 1;          // 0..3 (32 rows each)
    const int wn    = wave & 1;           // 0..1 (32 cols each)
    const int dTile = blockIdx.x;         // 0..7
    const int kTile = blockIdx.y;         // 0..7
    const int bb    = blockIdx.z;         // batch

    const _Float16* Wt[2] = {Wc, Wm};
    const float*    Ts[2] = {tsa, tsb};

    const int l15 = lane & 15;
    const int lhi = lane >> 4;

    v8f acc[2][2][2][2];                  // [tensor][part][m16][n16]
    const v8f vzero = {};
#pragma unroll
    for (int e = 0; e < 2; ++e)
#pragma unroll
        for (int p = 0; p < 2; ++p)
#pragma unroll
            for (int m = 0; m < 2; ++m)
#pragma unroll
                for (int n = 0; n < 2; ++n) acc[e][p][m][n] = vzero;

    for (int k0 = 0; k0 < TLEN; k0 += BK) {
        // --- W tiles: 2 tables x 128 rows, one 64B row per thread ----------
        {
            const int p  = tid >> 7;
            const int lr = tid & 127;
            const uint4* src =
                (const uint4*)(Wt[p] + (size_t)(kTile * BM + lr) * TLEN + k0);
            uint4* dst = (uint4*)(&As[p][lr][0]);
            dst[0] = src[0]; dst[1] = src[1]; dst[2] = src[2]; dst[3] = src[3];
        }
        // --- ts tiles: f32 load (coalesced along d), f16 + transpose store -
#pragma unroll
        for (int i = 0; i < 4; ++i) {
            const int c  = tid + i * 256;           // 1024 float4 chunks
            const int e  = c >> 9;                  // tensor
            const int r  = c & 511;
            const int tt = r >> 4;                  // 0..31  (time within BK)
            const int dc = r & 15;                  // float4 chunk along d
            const float4 v = *(const float4*)(Ts[e] + (size_t)bb * TLEN * DDIM +
                                              (size_t)(k0 + tt) * DDIM +
                                              dTile * BN + dc * 4);
            Bs[e][dc * 4 + 0][tt] = (_Float16)v.x;
            Bs[e][dc * 4 + 1][tt] = (_Float16)v.y;
            Bs[e][dc * 4 + 2][tt] = (_Float16)v.z;
            Bs[e][dc * 4 + 3][tt] = (_Float16)v.w;
        }
        __syncthreads();

        // --- fragments (layouts per ISA 7.12.2, 16-bit A 16x32 / B 32x16) --
        v16h af[2][2];
#pragma unroll
        for (int p = 0; p < 2; ++p)
#pragma unroll
            for (int m16 = 0; m16 < 2; ++m16) {
                const int row  = wm * 32 + m16 * 16 + l15;
                const int koff = lhi * 8;
                v8h lo = *(const v8h*)&As[p][row][koff];
                v8h hi = *(const v8h*)&As[p][row][16 + koff];
                af[p][m16] = __builtin_shufflevector(
                    lo, hi, 0, 1, 2, 3, 4, 5, 6, 7, 8, 9, 10, 11, 12, 13, 14, 15);
            }
        v16h bf[2][2];
#pragma unroll
        for (int e = 0; e < 2; ++e)
#pragma unroll
            for (int n16 = 0; n16 < 2; ++n16) {
                const int n  = wn * 32 + n16 * 16 + l15;
                const int kb = lhi * 8;
                v8h lo = *(const v8h*)&Bs[e][n][kb];
                v8h hi = *(const v8h*)&Bs[e][n][16 + kb];
                bf[e][n16] = __builtin_shufflevector(
                    lo, hi, 0, 1, 2, 3, 4, 5, 6, 7, 8, 9, 10, 11, 12, 13, 14, 15);
            }
        // --- 16 WMMAs: every A frag x every B frag -------------------------
#pragma unroll
        for (int e = 0; e < 2; ++e)
#pragma unroll
            for (int p = 0; p < 2; ++p)
#pragma unroll
                for (int m16 = 0; m16 < 2; ++m16)
#pragma unroll
                    for (int n16 = 0; n16 < 2; ++n16)
                        acc[e][p][m16][n16] =
                            __builtin_amdgcn_wmma_f32_16x16x32_f16(
                                false, af[p][m16], false, bf[e][n16],
                                (short)0, acc[e][p][m16][n16], false, false);
        __syncthreads();
    }

    // --- epilogue: reduce over d (cols), atomic-accumulate per (b,k) -------
    // C frag layout: elem r at lane l -> row r + 8*(l>>4), col l&15.
#pragma unroll
    for (int m16 = 0; m16 < 2; ++m16) {
        v8f da = vzero, db = vzero, cr = vzero, ci = vzero;
#pragma unroll
        for (int n16 = 0; n16 < 2; ++n16) {
            v8f ar = acc[0][0][m16][n16];   // Re Qa
            v8f ai = acc[0][1][m16][n16];   // Im Qa
            v8f br = acc[1][0][m16][n16];   // Re Qb
            v8f bi = acc[1][1][m16][n16];   // Im Qb
            da += ar * ar + ai * ai;
            db += br * br + bi * bi;
            cr += ar * br + ai * bi;        // Re(conj(Qa)*Qb)
            ci += ar * bi - ai * br;        // Im(conj(Qa)*Qb)
        }
        // butterfly over the 16-lane halves (sums the 16 d-columns)
#pragma unroll
        for (int msk = 1; msk <= 8; msk <<= 1) {
#pragma unroll
            for (int r = 0; r < 8; ++r) {
                da[r] += __shfl_xor(da[r], msk, 32);
                db[r] += __shfl_xor(db[r], msk, 32);
                cr[r] += __shfl_xor(cr[r], msk, 32);
                ci[r] += __shfl_xor(ci[r], msk, 32);
            }
        }
        if (l15 == 0) {  // lanes 0 (rows 0..7) and 16 (rows 8..15)
            const int kg = kTile * BM + wm * 32 + m16 * 16 + lhi * 8;
            float* base = accg + (size_t)bb * 4 * NBIN;
#pragma unroll
            for (int r = 0; r < 8; ++r) {
                atomicAdd(base + 0 * NBIN + kg + r, da[r]);
                atomicAdd(base + 1 * NBIN + kg + r, db[r]);
                atomicAdd(base + 2 * NBIN + kg + r, cr[r]);
                atomicAdd(base + 3 * NBIN + kg + r, ci[r]);
            }
        }
    }
}

// ------------------ inverse DFT (1024 bins -> 4096) + concat ---------------
__global__ __launch_bounds__(256)
void cte_idft_kernel(const float* __restrict__ accg, float* __restrict__ out) {
    __shared__ float Sr[NBIN];
    __shared__ float Si[NBIN];
    const int tid   = threadIdx.x;
    const int chunk = blockIdx.x;   // 0..15 (256 n each)
    const int dir   = blockIdx.y;   // 0: corr(Qa,Qb), 1: corr(Qb,Qa)
    const int b     = blockIdx.z;

    const float* base = accg + (size_t)b * 4 * NBIN;
    for (int k = tid; k < NBIN; k += 256) {
        const float den = base[(dir == 0 ? 0 : 1) * NBIN + k] + LMBDA;
        const float crv = base[2 * NBIN + k];
        const float civ = base[3 * NBIN + k];
        Sr[k] = crv / den;
        Si[k] = (dir == 0 ? civ : -civ) / den;   // Sba = conj(cross)/denb
    }
    __syncthreads();

    const int n = chunk * 256 + tid;             // 0..4095
    const float w = TWO_PI / (float)TLEN;
    float cd, sd;
    __sincosf(w * (float)(n & (TLEN - 1)), &sd, &cd);  // per-k rotation
    float sum = 0.0f;
#pragma unroll 1
    for (int blk = 0; blk < NBIN / 128; ++blk) {
        const int k0 = blk * 128;
        float c, s;                               // exact resync each block
        __sincosf(w * (float)((k0 * n) & (TLEN - 1)), &s, &c);
#pragma unroll 4
        for (int kk = 0; kk < 128; ++kk) {
            const int k = k0 + kk;
            sum = fmaf(Sr[k], c, sum);
            sum = fmaf(-Si[k], s, sum);           // + Sr*cos - Si*sin
            const float nc = fmaf(c, cd, -s * sd);
            const float ns = fmaf(s, cd, c * sd);
            c = nc; s = ns;
        }
    }
    sum *= (1.0f / (float)TLEN);                  // normalized ifft

    if (dir == 0)
        out[(size_t)b * (2 * TLEN) + (TLEN - 1 - n)] = sum;  // flipped half
    else
        out[(size_t)b * (2 * TLEN) + TLEN + n] = sum;
}

// ---------------------------------------------------------------------------
extern "C" void kernel_launch(void* const* d_in, const int* in_sizes, int n_in,
                              void* d_out, int out_size, void* d_ws, size_t ws_size,
                              hipStream_t stream) {
    const float* tsa = (const float*)d_in[0];
    const float* tsb = (const float*)d_in[1];
    // d_in[2] (offsets) only determines max_len == 4096; data unused.
    float* out = (float*)d_out;

    char* ws = (char*)d_ws;
    _Float16* Wc   = (_Float16*)(ws);
    _Float16* Wm   = (_Float16*)(ws + WS_TBL_BYTES);
    float*    accg = (float*)   (ws + 2 * WS_TBL_BYTES);

    // 1) twiddle tables (rebuilt every call; deterministic)
    cte_table_kernel<<<(NBIN * TLEN) / 256, 256, 0, stream>>>(Wc, Wm);
    // 2) zero the (poisoned) per-(b,k) accumulators
    cte_zero_kernel<<<(BATCH * 4 * NBIN + 255) / 256, 256, 0, stream>>>(
        accg, BATCH * 4 * NBIN);
    // 3) fused WMMA DFT + d-reduction
    cte_dft_corr_kernel<<<dim3(DDIM / 64, NBIN / 128, BATCH), 256, 0, stream>>>(
        tsa, tsb, Wc, Wm, accg);
    // 4) inverse DFT, flip, concat
    cte_idft_kernel<<<dim3(TLEN / 256, 2, BATCH), 256, 0, stream>>>(accg, out);
}
